// NDNRefinement_60129542609
// MI455X (gfx1250) — compile-verified, matrix operand order checked by reading, not run on hardware
//
#include <hip/hip_runtime.h>

// ---------------------------------------------------------------------------
// Types
// ---------------------------------------------------------------------------
typedef __attribute__((ext_vector_type(16))) __bf16 bf16x16;
typedef __attribute__((ext_vector_type(8)))  float  f32x8;
typedef __attribute__((ext_vector_type(4)))  float  f32x4;
typedef __attribute__((ext_vector_type(4)))  int    i32x4;

#define BLK_MN 128
#define KSTEP  32
#define LDSS   40   // padded LDS row stride in bf16 elements (80B, 16B-aligned, conflict-free)

enum { ASRC_BF16 = 0, ASRC_TRIPLE = 1, ASRC_EMB = 2, ASRC_POOLED = 3 };
enum { EPI_BF16 = 0, EPI_F32 = 1, EPI_SCATTER = 2 };

struct GemmP {
  // A sources
  const unsigned short* Abf; int lda;                 // ASRC_BF16  (K multiple of 32)
  const unsigned short* obj; int obj_ld;              // ASRC_TRIPLE (din mult of 16)
  const unsigned short* pred; int pred_ld;
  const int* sidx; const int* oidx; int din;
  const float* objv; const float* boxes;              // ASRC_EMB
  const float* pooledA; const float* counts;          // ASRC_POOLED (stride == K, K mult of 32)
  // B (K x N row-major, bf16) and bias (f32, length N)
  const unsigned short* Bw; const float* bias;
  // outputs
  unsigned short* Dbf; float* Df32; int ldd;          // EPI_BF16 / EPI_F32
  float* pooled_out; unsigned short* pred_out;        // EPI_SCATTER
  int h; int dout;
  int M, N, K;
};

__device__ __forceinline__ unsigned short f2bf(float f) {
  unsigned u = __float_as_uint(f);
  unsigned r = ((u >> 16) & 1u) + 0x7FFFu;            // round-to-nearest-even
  return (unsigned short)((u + r) >> 16);
}

// Load one 16-row slice of a 16x32 bf16 fragment from LDS per the CDNA5 16-bit
// A/B layout: lane half h: dwords 0-3 = K[8h..8h+7], dwords 4-7 = K[16+8h..16+8h+7]
__device__ __forceinline__ bf16x16 ldsfrag(const unsigned short* base, int row, int khalf) {
  const i32x4* p0 = reinterpret_cast<const i32x4*>(base + row * LDSS + khalf);
  const i32x4* p1 = reinterpret_cast<const i32x4*>(base + row * LDSS + 16 + khalf);
  union { i32x4 q[2]; bf16x16 v; } u;
  u.q[0] = *p0;
  u.q[1] = *p1;
  return u.v;
}

// One K-slab staged in registers: A 2x16B + B 2x16B per thread.
struct Slab { i32x4 a0, a1, b0, b1; };

// Branchless, vectorized staging of one K-slab (A: 128x32, B^T: 128x32).
// Thread t owns row/col r = t>>1, k-run kseg = k0 + (t&1)*16 .. +15.
// All segment/K boundaries for vector paths are multiples of 16, so each
// thread's 16-element run is segment-uniform -> unconditional b128 loads.
template <int ASRC>
__device__ __forceinline__ Slab stage_load(const GemmP& p, int m0, int n0, int k0,
                                           const int* sS, const int* sO) {
  const int tid = threadIdx.x;
  const int r    = tid >> 1;
  const int kseg = k0 + (tid & 1) * 16;
  const int row  = m0 + r;
  const bool rowok = (row < p.M);
  const long long rowc = rowok ? row : 0;     // clamped for safe addresses
  const i32x4 zz = {0, 0, 0, 0};
  Slab s;

  // ---- A slab ----
  if (ASRC == ASRC_BF16) {
    const i32x4* av = reinterpret_cast<const i32x4*>(p.Abf + rowc * p.lda + kseg);
    s.a0 = av[0]; s.a1 = av[1];
  } else if (ASRC == ASRC_TRIPLE) {
    const unsigned short* base;
    if (kseg < p.din)          base = p.obj  + (long long)sS[r] * p.obj_ld + kseg;
    else if (kseg < 2 * p.din) base = p.pred + rowc * p.pred_ld + (kseg - p.din);
    else                       base = p.obj  + (long long)sO[r] * p.obj_ld + (kseg - 2 * p.din);
    const i32x4* av = reinterpret_cast<const i32x4*>(base);
    s.a0 = av[0]; s.a1 = av[1];
  } else if (ASRC == ASRC_EMB) {
    union { unsigned short u[16]; i32x4 q[2]; } ua;
    if (kseg + 16 <= 64) {           // fully inside obj_vecs: 4x b128 f32
      const f32x4* ov = reinterpret_cast<const f32x4*>(p.objv + rowc * 64 + kseg);
      f32x4 f0 = ov[0], f1 = ov[1], f2 = ov[2], f3 = ov[3];
#pragma unroll
      for (int i = 0; i < 4; i++) {
        ua.u[i]      = f2bf(f0[i]);
        ua.u[4 + i]  = f2bf(f1[i]);
        ua.u[8 + i]  = f2bf(f2[i]);
        ua.u[12 + i] = f2bf(f3[i]);
      }
    } else if (kseg < 68) {          // boxes segment (k = 64..67) + zero pad
      const float* bx = p.boxes + rowc * 4;
      f32x4 b = *reinterpret_cast<const f32x4*>(bx);
#pragma unroll
      for (int i = 0; i < 16; i++) {
        const int k = kseg + i;
        ua.u[i] = (k < 68) ? f2bf(b[k - 64]) : (unsigned short)0;
      }
    } else {
#pragma unroll
      for (int i = 0; i < 16; i++) ua.u[i] = 0;
    }
    s.a0 = ua.q[0]; s.a1 = ua.q[1];
  } else { // ASRC_POOLED: contiguous f32 row, normalize by clamped counts
    const f32x4* pv = reinterpret_cast<const f32x4*>(p.pooledA + rowc * p.K + kseg);
    f32x4 f0 = pv[0], f1 = pv[1], f2 = pv[2], f3 = pv[3];
    float c = p.counts[rowc];
    float rc = 1.0f / ((c > 1.0f) ? c : 1.0f);
    union { unsigned short u[16]; i32x4 q[2]; } ua;
#pragma unroll
    for (int i = 0; i < 4; i++) {
      ua.u[i]      = f2bf(f0[i] * rc);
      ua.u[4 + i]  = f2bf(f1[i] * rc);
      ua.u[8 + i]  = f2bf(f2[i] * rc);
      ua.u[12 + i] = f2bf(f3[i] * rc);
    }
    s.a0 = ua.q[0]; s.a1 = ua.q[1];
  }
  if (!rowok) { s.a0 = zz; s.a1 = zz; }       // M-tail: zero-select, no branchy loads

  // ---- B slab: br[kk] = B[kseg+kk][n0+r], clamped addrs -> unconditional loads ----
  {
    const int ng   = n0 + r;
    const bool ngok = (ng < p.N);
    const int ngc  = ngok ? ng : (p.N - 1);
    const unsigned short* bp = p.Bw + ngc;
    union { unsigned short u[16]; i32x4 q[2]; } ub;
#pragma unroll
    for (int kk = 0; kk < 16; kk++) {
      const int k  = kseg + kk;
      const int kc = (k < p.K) ? k : (p.K - 1);
      unsigned short v = bp[(long long)kc * p.N];
      ub.u[kk] = (ngok && k < p.K) ? v : (unsigned short)0;
    }
    s.b0 = ub.q[0]; s.b1 = ub.q[1];
  }
  return s;
}

// Commit staged registers to an LDS ping-pong buffer with 2x b128 stores each.
__device__ __forceinline__ void stage_store(unsigned short* lAbuf, unsigned short* lBbuf,
                                            const Slab& s) {
  const int tid = threadIdx.x;
  const int r   = tid >> 1;
  const int kb  = (tid & 1) * 16;
  i32x4* pa = reinterpret_cast<i32x4*>(lAbuf + r * LDSS + kb);
  pa[0] = s.a0; pa[1] = s.a1;
  i32x4* pb = reinterpret_cast<i32x4*>(lBbuf + r * LDSS + kb);
  pb[0] = s.b0; pb[1] = s.b1;
}

// ---------------------------------------------------------------------------
// Fused WMMA GEMM: D = epilogue(leaky(A @ B + bias))
// 128x128 block tile, 8 wave32s, double-buffered LDS, software-pipelined
// global->reg->LDS staging overlapped with batched ds_load_b128 + 8x WMMA.
// ---------------------------------------------------------------------------
template <int ASRC, int EPI>
__global__ void __launch_bounds__(256) gemm_kernel(GemmP p) {
  __shared__ unsigned short lA[2][BLK_MN * LDSS];
  __shared__ unsigned short lB[2][BLK_MN * LDSS];
  __shared__ int sS[BLK_MN], sO[BLK_MN];

  const int tid  = threadIdx.x;
  const int wave = tid >> 5;
  const int lane = tid & 31;
  const int m0 = blockIdx.y * BLK_MN;
  const int n0 = blockIdx.x * BLK_MN;

  if (ASRC == ASRC_TRIPLE) {
    for (int r = tid; r < BLK_MN; r += 256) {
      int row = m0 + r;
      int s = 0, o = 0;
      if (row < p.M) { s = p.sidx[row]; o = p.oidx[row]; }
      sS[r] = s; sO[r] = o;
    }
    __syncthreads();
  }

  f32x8 acc[8];
#pragma unroll
  for (int n = 0; n < 8; n++)
#pragma unroll
    for (int j = 0; j < 8; j++) acc[n][j] = 0.0f;

  const int ksteps = (p.K + KSTEP - 1) / KSTEP;

  // Prologue: stage K-slab 0 into buffer 0.
  Slab sl = stage_load<ASRC>(p, m0, n0, 0, sS, sO);
  stage_store(lA[0], lB[0], sl);
  __syncthreads();

  const int rr    = lane & 15;
  const int khalf = (lane >> 4) * 8;

  for (int ks = 0; ks < ksteps; ks++) {
    const int cur = ks & 1, nxt = cur ^ 1;
    const bool more = (ks + 1 < ksteps);

    if (more) {  // prefetch the slab after next (global_prefetch_b8)
      __builtin_prefetch(p.Bw + (size_t)(ks * KSTEP + 2 * KSTEP) * p.N + n0 + (tid & 127), 0, 1);
    }

    // Batched fragment loads: 18x ds_load_b128, staggered waits, 8 WMMAs back-to-back.
    bf16x16 af = ldsfrag(lA[cur], wave * 16 + rr, khalf);
    bf16x16 bfv[8];
#pragma unroll
    for (int n = 0; n < 8; n++) bfv[n] = ldsfrag(lB[cur], n * 16 + rr, khalf);

    // Issue next slab's global loads now so they overlap the WMMAs below.
    if (more) sl = stage_load<ASRC>(p, m0, n0, (ks + 1) * KSTEP, sS, sO);

#pragma unroll
    for (int n = 0; n < 8; n++) {
      acc[n] = __builtin_amdgcn_wmma_f32_16x16x32_bf16(
          false, af, false, bfv[n], (short)0, acc[n], false, false);
    }

    if (more) stage_store(lA[nxt], lB[nxt], sl);
    __syncthreads();
  }

  // ---- epilogue: C layout is VGPR j, lanes0-15 -> M=j, lanes16-31 -> M=j+8 ----
  const int cl   = lane & 15;
  const int half = lane >> 4;
  float bb[8];
#pragma unroll
  for (int n = 0; n < 8; n++) {
    const int col = n0 + n * 16 + cl;
    bb[n] = (col < p.N) ? p.bias[col] : 0.0f;
  }
#pragma unroll
  for (int j = 0; j < 8; j++) {
    const int row = m0 + wave * 16 + j + 8 * half;
    if (row >= p.M) continue;
    int si = 0, oi = 0;
    if (EPI == EPI_SCATTER) { si = p.sidx[row]; oi = p.oidx[row]; }
#pragma unroll
    for (int n = 0; n < 8; n++) {
      const int col = n0 + n * 16 + cl;
      if (col >= p.N) continue;
      float v = acc[n][j] + bb[n];
      v = (v > 0.0f) ? v : 0.2f * v;                  // leaky(0.2)
      if (EPI == EPI_BF16) {
        p.Dbf[(long long)row * p.ldd + col] = f2bf(v);
      } else if (EPI == EPI_F32) {
        p.Df32[(long long)row * p.ldd + col] = v;
      } else { // EPI_SCATTER: new_s | new_p | new_o column split
        if (col < p.h) {
          atomicAdd(&p.pooled_out[(long long)si * p.h + col], v);
        } else if (col < p.h + p.dout) {
          p.pred_out[(long long)row * p.dout + (col - p.h)] = f2bf(v);
        } else {
          atomicAdd(&p.pooled_out[(long long)oi * p.h + (col - p.h - p.dout)], v);
        }
      }
    }
  }
}

// ---------------------------------------------------------------------------
// Aux kernels
// ---------------------------------------------------------------------------
__global__ void cvt_kernel(const float* __restrict__ src, unsigned short* __restrict__ dst,
                           long long n) {
  long long i = (long long)blockIdx.x * blockDim.x + threadIdx.x;
  if (i < n) dst[i] = f2bf(src[i]);
}

__global__ void zero_kernel(float* dst, long long n) {
  long long i = (long long)blockIdx.x * blockDim.x + threadIdx.x;
  if (i < n) dst[i] = 0.0f;
}

__global__ void counts_kernel(const int* __restrict__ s, const int* __restrict__ o,
                              float* __restrict__ c, int T) {
  int i = blockIdx.x * blockDim.x + threadIdx.x;
  if (i < T) {
    atomicAdd(&c[s[i]], 1.0f);
    atomicAdd(&c[o[i]], 1.0f);
  }
}

template <int ASRC, int EPI>
static void launch_gemm(const GemmP& p, hipStream_t s) {
  dim3 g((p.N + BLK_MN - 1) / BLK_MN, (p.M + BLK_MN - 1) / BLK_MN);
  gemm_kernel<ASRC, EPI><<<g, dim3(256), 0, s>>>(p);
}

// ---------------------------------------------------------------------------
// Host orchestration
// ---------------------------------------------------------------------------
extern "C" void kernel_launch(void* const* d_in, const int* in_sizes, int n_in,
                              void* d_out, int out_size, void* d_ws, size_t ws_size,
                              hipStream_t stream) {
  (void)n_in; (void)out_size; (void)ws_size;

  const int O = in_sizes[0] / 64;
  const int T = in_sizes[3];

  const float* obj_vecs  = (const float*)d_in[0];
  const float* pred_vecs = (const float*)d_in[1];
  const float* boxes     = (const float*)d_in[2];
  const int*   sidx      = (const int*)d_in[3];
  const int*   oidx      = (const int*)d_in[4];
  const float* W_emb     = (const float*)d_in[5];
  const float* b_emb     = (const float*)d_in[6];
  const float* W_bb      = (const float*)d_in[39];
  const float* b_bb      = (const float*)d_in[40];

  static const int dims[4][3] = {{64,512,128},{128,512,128},{128,512,128},{128,128,128}};

  size_t off = 0;
  auto take = [&](size_t bytes) -> char* {
    size_t a = (off + 255) & ~(size_t)255;
    off = a + bytes;
    return (char*)d_ws + a;
  };

  unsigned short* obj_bf  = (unsigned short*)take((size_t)O * 128 * 2);
  unsigned short* pred_bf = (unsigned short*)take((size_t)T * 128 * 2);
  unsigned short* hid_bf  = (unsigned short*)take((size_t)T * 512 * 2);
  unsigned short* tmp_bf  = (unsigned short*)take((size_t)O * 512 * 2);
  float*          pooled  = (float*)take((size_t)O * 512 * 4);
  float*          counts  = (float*)take((size_t)O * 4);

  auto cvtw = [&](const void* src, size_t n) -> unsigned short* {
    unsigned short* dst = (unsigned short*)take(n * 2);
    cvt_kernel<<<dim3((unsigned)((n + 255) / 256)), dim3(256), 0, stream>>>(
        (const float*)src, dst, (long long)n);
    return dst;
  };

  // Convert all weights to bf16 (biases stay f32)
  unsigned short* wEmb = cvtw(W_emb, (size_t)68 * 64);
  unsigned short *w1a[4], *w1b[4], *w2a[4], *w2b[4];
  const float *b1a[4], *b1b[4], *b2a[4], *b2b[4];
  for (int i = 0; i < 4; i++) {
    const int din = dims[i][0], h = dims[i][1], dout = dims[i][2];
    const int base = 7 + i * 8;
    w1a[i] = cvtw(d_in[base + 0], (size_t)3 * din * h);
    b1a[i] = (const float*)d_in[base + 1];
    w1b[i] = cvtw(d_in[base + 2], (size_t)h * (2 * h + dout));
    b1b[i] = (const float*)d_in[base + 3];
    w2a[i] = cvtw(d_in[base + 4], (size_t)h * h);
    b2a[i] = (const float*)d_in[base + 5];
    w2b[i] = cvtw(d_in[base + 6], (size_t)h * dout);
    b2b[i] = (const float*)d_in[base + 7];
  }
  unsigned short* wBB = cvtw(W_bb, (size_t)128 * 4);

  // Segment counts (f32 atomics)
  zero_kernel<<<dim3((O + 255) / 256), dim3(256), 0, stream>>>(counts, (long long)O);
  counts_kernel<<<dim3((T + 255) / 256), dim3(256), 0, stream>>>(sidx, oidx, counts, T);

  // Initial pred in bf16
  cvt_kernel<<<dim3((unsigned)(((size_t)T * 64 + 255) / 256)), dim3(256), 0, stream>>>(
      pred_vecs, pred_bf, (long long)T * 64);

  // Embedding: obj = leaky([obj_vecs | boxes] @ W_emb + b_emb)   (O x 68) @ (68 x 64)
  {
    GemmP p{};
    p.objv = obj_vecs; p.boxes = boxes;
    p.Bw = wEmb; p.bias = b_emb;
    p.Dbf = obj_bf; p.ldd = 64;
    p.M = O; p.N = 64; p.K = 68;
    launch_gemm<ASRC_EMB, EPI_BF16>(p, stream);
  }

  int obj_ld = 64, pred_ld = 64;
  for (int i = 0; i < 4; i++) {
    const int din = dims[i][0], h = dims[i][1], dout = dims[i][2];

    // gemm1: hid = leaky([obj[s]|pred|obj[o]] @ W1a + b1a)   (T x 3din) @ (3din x h)
    {
      GemmP p{};
      p.obj = obj_bf; p.obj_ld = obj_ld;
      p.pred = pred_bf; p.pred_ld = pred_ld;
      p.sidx = sidx; p.oidx = oidx; p.din = din;
      p.Bw = w1a[i]; p.bias = b1a[i];
      p.Dbf = hid_bf; p.ldd = h;
      p.M = T; p.N = h; p.K = 3 * din;
      launch_gemm<ASRC_TRIPLE, EPI_BF16>(p, stream);
    }

    zero_kernel<<<dim3((unsigned)(((size_t)O * h + 255) / 256)), dim3(256), 0, stream>>>(
        pooled, (long long)O * h);

    // gemm2 + split/scatter: out = leaky(hid @ W1b + b1b); new_s/new_o -> pooled, new_p -> pred
    {
      GemmP p{};
      p.Abf = hid_bf; p.lda = h;
      p.sidx = sidx; p.oidx = oidx;
      p.Bw = w1b[i]; p.bias = b1b[i];
      p.pooled_out = pooled; p.pred_out = pred_bf; p.h = h; p.dout = dout;
      p.M = T; p.N = 2 * h + dout; p.K = h;
      launch_gemm<ASRC_BF16, EPI_SCATTER>(p, stream);
    }

    // obj MLP a: tmp = leaky((pooled / counts) @ W2a + b2a)   (O x h) @ (h x h)
    {
      GemmP p{};
      p.pooledA = pooled; p.counts = counts;
      p.Bw = w2a[i]; p.bias = b2a[i];
      p.Dbf = tmp_bf; p.ldd = h;
      p.M = O; p.N = h; p.K = h;
      launch_gemm<ASRC_POOLED, EPI_BF16>(p, stream);
    }

    // obj MLP b: obj = leaky(tmp @ W2b + b2b)   (O x h) @ (h x dout)
    {
      GemmP p{};
      p.Abf = tmp_bf; p.lda = h;
      p.Bw = w2b[i]; p.bias = b2b[i];
      p.Dbf = obj_bf; p.ldd = dout;
      p.M = O; p.N = dout; p.K = h;
      launch_gemm<ASRC_BF16, EPI_BF16>(p, stream);
    }

    obj_ld = dout; pred_ld = dout;
  }

  // Final bbox head: out = leaky(obj @ W_bb + b_bb)   (O x 128) @ (128 x 4), f32 output
  {
    GemmP p{};
    p.Abf = obj_bf; p.lda = 128;
    p.Bw = wBB; p.bias = b_bb;
    p.Df32 = (float*)d_out; p.ldd = 4;
    p.M = O; p.N = 4; p.K = 128;
    launch_gemm<ASRC_BF16, EPI_F32>(p, stream);
  }
}
